// BiRNN1_27324581937388
// MI455X (gfx1250) — matrix operand order repaired and checked by reading, not compiled, not run on hardware
//
#include <hip/hip_runtime.h>
#include <hip/hip_bf16.h>
#include <math.h>

typedef __attribute__((ext_vector_type(16))) _Float16 v16h;
typedef __attribute__((ext_vector_type(8)))  float    v8f;
typedef __attribute__((ext_vector_type(2)))  __fp16   fp16x2;  // cvt_pkrtz result type

#define S_LEN 1024
#define BATCH 256
#define DIN   64
#define HID   128

__device__ __forceinline__ float fast_tanh(float x) {
#if __has_builtin(__builtin_amdgcn_tanhf)
    return __builtin_amdgcn_tanhf(x);
#else
    return tanhf(x);
#endif
}

__device__ __forceinline__ v8f wmma_f16(v16h a, v16h b, v8f c) {
    // D = A(16x32 f16) * B(32x16 f16) + C(16x16 f32)
    return __builtin_amdgcn_wmma_f32_16x16x32_f16(
        /*neg_a=*/false, a, /*neg_b=*/false, b,
        /*c_mod=*/(short)0, c, /*reuse_a=*/false, /*reuse_b=*/false);
}

// One workgroup = (batch tile of 16, direction). 4 waves, each owns 2 M-tiles
// (32 h_out rows). GEMM orientation: h^T_new = tanh(Whh * h^T + Wih * x^T + b),
// tiles are (M=h_out, N=batch16, K=h_in/d). A-fragments (weights) live in
// VGPRs for the whole scan; hidden state ping-pongs through LDS in [b][h] f16
// layout, which is exactly per-lane-contiguous for B-fragment reads.
__global__ __launch_bounds__(128)
void birnn_scan(const float* __restrict__ x,
                const float* __restrict__ Wih_fw, const float* __restrict__ Whh_fw,
                const float* __restrict__ bih_fw, const float* __restrict__ bhh_fw,
                const float* __restrict__ Wih_bw, const float* __restrict__ Whh_bw,
                const float* __restrict__ bih_bw, const float* __restrict__ bhh_bw,
                const float* __restrict__ fcW,
                float* __restrict__ partial)
{
    const int bt   = blockIdx.x;       // batch tile 0..15
    const int dir  = blockIdx.y;       // 0 = fw, 1 = bw
    const int tid  = threadIdx.x;
    const int w    = tid >> 5;         // wave id 0..3
    const int lane = tid & 31;
    const int lo   = lane & 15;        // N (batch row in tile) / M row for A
    const int hi   = (lane >> 4) & 1;  // lane-half selector

    const float* Wih = dir ? Wih_bw : Wih_fw;
    const float* Whh = dir ? Whh_bw : Whh_fw;
    const float* bih = dir ? bih_bw : bih_fw;
    const float* bhh = dir ? bhh_bw : bhh_fw;

    __shared__ _Float16 hbuf[2][16 * HID];   // ping-pong h^T as [b'][h], f16

    for (int idx = tid; idx < 16 * HID; idx += 128)
        hbuf[0][idx] = (_Float16)0.f;

    // ---- Preload constant A-fragments, bias, fc weights (once per scan) ----
    v16h whhA[2][4];   // 2 M-tiles x 4 K-chunks over h_in (K=32 each)
    v16h wihA[2][2];   // 2 M-tiles x 2 K-chunks over d    (K=32 each)
    v8f  biasv[2];
    float fcw[2][8];
#pragma unroll
    for (int m = 0; m < 2; ++m) {
        const int mbase = (w * 2 + m) * 16;
        const int row   = mbase + lo;          // A-matrix: M = lane%16
#pragma unroll
        for (int kc = 0; kc < 4; ++kc) {
            v16h f;
#pragma unroll
            for (int e = 0; e < 16; ++e) {
                const int k = kc * 32 + hi * 16 + e;
                f[e] = (_Float16)Whh[row * HID + k];
            }
            whhA[m][kc] = f;
        }
#pragma unroll
        for (int kc = 0; kc < 2; ++kc) {
            v16h f;
#pragma unroll
            for (int e = 0; e < 16; ++e) {
                const int k = kc * 32 + hi * 16 + e;
                f[e] = (_Float16)Wih[row * DIN + k];
            }
            wihA[m][kc] = f;
        }
#pragma unroll
        for (int r = 0; r < 8; ++r) {
            const int h = mbase + r + hi * 8;  // D layout: M = r + 8*hi
            biasv[m][r] = bih[h] + bhh[h];
            fcw[m][r]   = fcW[dir * HID + h];
        }
    }
    __syncthreads();

    // x^T B-fragment source: per lane 16 consecutive d at fixed batch row
    const float* xbase = x + (size_t)(bt * 16 + lo) * (S_LEN * DIN) + hi * 16;

    auto load_x = [&](int t, float4 (&buf)[2][4]) {
#pragma unroll
        for (int c = 0; c < 2; ++c)
#pragma unroll
            for (int j = 0; j < 4; ++j)
                buf[c][j] = *reinterpret_cast<const float4*>(
                    xbase + t * DIN + c * 32 + j * 4);
    };
    auto cvt_x = [&](const float4 (&buf)[2][4], v16h (&xf)[2]) {
#pragma unroll
        for (int c = 0; c < 2; ++c) {
            v16h f;
            f[0]  = (_Float16)buf[c][0].x; f[1]  = (_Float16)buf[c][0].y;
            f[2]  = (_Float16)buf[c][0].z; f[3]  = (_Float16)buf[c][0].w;
            f[4]  = (_Float16)buf[c][1].x; f[5]  = (_Float16)buf[c][1].y;
            f[6]  = (_Float16)buf[c][1].z; f[7]  = (_Float16)buf[c][1].w;
            f[8]  = (_Float16)buf[c][2].x; f[9]  = (_Float16)buf[c][2].y;
            f[10] = (_Float16)buf[c][2].z; f[11] = (_Float16)buf[c][2].w;
            f[12] = (_Float16)buf[c][3].x; f[13] = (_Float16)buf[c][3].y;
            f[14] = (_Float16)buf[c][3].z; f[15] = (_Float16)buf[c][3].w;
            xf[c] = f;
        }
    };

    float4 xraw[2][4];
    v16h   xf[2];
    load_x(dir ? (S_LEN - 1) : 0, xraw);
    cvt_x(xraw, xf);

    float* pbase = partial + (size_t)(dir * 16 + bt) * S_LEN * 64 + w * 16;

    for (int i = 0; i < S_LEN; ++i) {
        const int t  = dir ? (S_LEN - 1 - i) : i;
        const int tn = dir ? (S_LEN - 2 - i) : (i + 1);

        // Prefetch next step's input fragments (independent of recurrence)
        float4 xrawn[2][4];
        if (i + 1 < S_LEN) load_x(tn, xrawn);

        // Fused input projection: acc = b + Wih * x^T
        v8f acc0 = biasv[0], acc1 = biasv[1];
        acc0 = wmma_f16(wihA[0][0], xf[0], acc0);
        acc1 = wmma_f16(wihA[1][0], xf[0], acc1);
        acc0 = wmma_f16(wihA[0][1], xf[1], acc0);
        acc1 = wmma_f16(wihA[1][1], xf[1], acc1);

        // Recurrent part: acc += Whh * h^T  (B-frags straight out of LDS)
        const _Float16* hb = &hbuf[i & 1][0];
#pragma unroll
        for (int kc = 0; kc < 4; ++kc) {
            v16h hB = *reinterpret_cast<const v16h*>(hb + lo * HID + kc * 32 + hi * 16);
            acc0 = wmma_f16(whhA[0][kc], hB, acc0);
            acc1 = wmma_f16(whhA[1][kc], hB, acc1);
        }

        // tanh, FC partial, and store new h (packed f16 pairs, adjacent h_out)
        float p = 0.f;
        _Float16* hn = &hbuf[(i + 1) & 1][0];
#pragma unroll
        for (int m = 0; m < 2; ++m) {
            const v8f a  = m ? acc1 : acc0;
            const int mb = (w * 2 + m) * 16 + hi * 8;
#pragma unroll
            for (int r = 0; r < 8; r += 2) {
                const float t0 = fast_tanh(a[r]);
                const float t1 = fast_tanh(a[r + 1]);
                p += t0 * fcw[m][r] + t1 * fcw[m][r + 1];
#if __has_builtin(__builtin_amdgcn_cvt_pkrtz)
                fp16x2 pk = __builtin_amdgcn_cvt_pkrtz(t0, t1);
                *reinterpret_cast<fp16x2*>(hn + lo * HID + mb + r) = pk;
#else
                hn[lo * HID + mb + r]     = (_Float16)t0;
                hn[lo * HID + mb + r + 1] = (_Float16)t1;
#endif
            }
        }
        // lanes L and L+16 share the same batch row: fold halves, store 16 rows
        p += __shfl_xor(p, 16, 32);
        if (lane < 16) pbase[(size_t)t * 64 + lane] = p;

        __syncthreads();   // publish h for the next step (ping-pong)

        if (i + 1 < S_LEN) cvt_x(xrawn, xf);
    }
}

__global__ void birnn_combine(const float* __restrict__ partial,
                              const float* __restrict__ fc_b,
                              float* __restrict__ out)
{
    const int id = blockIdx.x * blockDim.x + threadIdx.x;
    if (id >= BATCH * S_LEN) return;
    const int b = id / S_LEN, t = id % S_LEN;
    const int bt = b >> 4, bp = b & 15;
    float s = fc_b[0];
#pragma unroll
    for (int d = 0; d < 2; ++d)
#pragma unroll
        for (int w = 0; w < 4; ++w)
            s += partial[(size_t)(d * 16 + bt) * S_LEN * 64 + (size_t)t * 64 + w * 16 + bp];
    out[id] = s;
}

extern "C" void kernel_launch(void* const* d_in, const int* in_sizes, int n_in,
                              void* d_out, int out_size, void* d_ws, size_t ws_size,
                              hipStream_t stream) {
    const float* x      = (const float*)d_in[0];
    const float* Wih_fw = (const float*)d_in[1];
    const float* Whh_fw = (const float*)d_in[2];
    const float* bih_fw = (const float*)d_in[3];
    const float* bhh_fw = (const float*)d_in[4];
    const float* Wih_bw = (const float*)d_in[5];
    const float* Whh_bw = (const float*)d_in[6];
    const float* bih_bw = (const float*)d_in[7];
    const float* bhh_bw = (const float*)d_in[8];
    const float* fcW    = (const float*)d_in[9];
    const float* fcb    = (const float*)d_in[10];

    float* partial = (float*)d_ws;  // [2][16][1024][4][16] f32 = 8 MB

    dim3 grid(16, 2);
    birnn_scan<<<grid, 128, 0, stream>>>(x, Wih_fw, Whh_fw, bih_fw, bhh_fw,
                                         Wih_bw, Whh_bw, bih_bw, bhh_bw,
                                         fcW, partial);

    const int n = BATCH * S_LEN;
    birnn_combine<<<(n + 255) / 256, 256, 0, stream>>>(partial, fcb, (float*)d_out);
}
